// myMultiheadAttention_89515708383249
// MI455X (gfx1250) — compile-verified, matrix operand order checked by reading, not run on hardware
//
#include <hip/hip_runtime.h>
#include <hip/hip_bf16.h>

#define LSEQ 2048
#define NB   4
#define EMB  1024
#define NH   16
#define HDIM 64
#define MROWS (LSEQ*NB)   // 8192
#define BH    (NB*NH)     // 64

typedef __attribute__((ext_vector_type(16))) __bf16 bf16x16;
typedef __attribute__((ext_vector_type(8)))  float  f32x8;

union FragU { bf16x16 v; uint4 q[2]; };

__device__ __forceinline__ unsigned short f2bf(float x) {
  unsigned u = __float_as_uint(x);
  u += 0x7FFFu + ((u >> 16) & 1u);          // round-to-nearest-even
  return (unsigned short)(u >> 16);
}

__device__ __forceinline__ f32x8 wmma_bf16(bf16x16 a, bf16x16 b, f32x8 c) {
  return __builtin_amdgcn_wmma_f32_16x16x32_bf16(false, a, false, b,
                                                 (short)0, c, false, false);
}

// Load one 16x32 (rows x K) bf16 operand fragment from a row-major matrix
// (global or LDS through generic pointer).
// Per CDNA5 16-bit A layout: lane<16 -> row=lane,   K{0..7,16..23}
//                            lane>=16 -> row=lane-16, K{8..15,24..31}
__device__ __forceinline__ bf16x16 load_frag_g(const unsigned short* p,
                                               int rowBase, int ld, int kBase) {
  int lane = threadIdx.x & 31;
  int row  = rowBase + (lane & 15);
  int koff = (lane >> 4) << 3;
  const unsigned short* r = p + (long)row * ld + kBase + koff;
  FragU f;
  f.q[0] = *reinterpret_cast<const uint4*>(r);
  f.q[1] = *reinterpret_cast<const uint4*>(r + 16);
  return f.v;
}

// Async global->LDS copy of 16 bytes (gfx1250, tracked by ASYNCcnt).
__device__ __forceinline__ void async_ld_b128(unsigned lds_off, const void* gaddr) {
  asm volatile("global_load_async_to_lds_b128 %0, %1, off"
               :: "v"(lds_off), "v"(gaddr) : "memory");
}

// ---------------------------------------------------------------- fp32 -> bf16
__global__ void cvt_f32_bf16(const float* __restrict__ in,
                             unsigned short* __restrict__ out, int n) {
  int i = blockIdx.x * blockDim.x + threadIdx.x;
  int stride = gridDim.x * blockDim.x;
  for (; i < n; i += stride) out[i] = f2bf(in[i]);
}

// ---------------------------------------------------------------- NT GEMM
// C[m,n] = sum_k A[m,k]*Bw[n,k] + bias[n];  M=8192, N=K=1024.
// mode 0: Q -> dQK [bh][l][hd] scaled by 1/sqrt(64)
// mode 1: K -> dQK [bh][l][hd]
// mode 2: V -> dQK [bh][hd][l]   (transposed for PV gemm)
// mode 3: fp32 -> dF row-major   (out-projection)
__global__ __launch_bounds__(256) void gemm_nt(
    const unsigned short* __restrict__ A,
    const unsigned short* __restrict__ Bw,
    const float* __restrict__ bias,
    unsigned short* __restrict__ dQK,
    float* __restrict__ dF,
    int mode) {
  int wid  = threadIdx.x >> 5;
  int lane = threadIdx.x & 31;
  int m0 = blockIdx.x * 64  + (wid >> 2) * 32;
  int n0 = blockIdx.y * 128 + (wid & 3)  * 32;

  f32x8 acc[2][2];
  acc[0][0] = {}; acc[0][1] = {}; acc[1][0] = {}; acc[1][1] = {};

  for (int k = 0; k < EMB; k += 32) {
    bf16x16 a0 = load_frag_g(A,  m0,      EMB, k);
    bf16x16 a1 = load_frag_g(A,  m0 + 16, EMB, k);
    bf16x16 b0 = load_frag_g(Bw, n0,      EMB, k);
    bf16x16 b1 = load_frag_g(Bw, n0 + 16, EMB, k);
    acc[0][0] = wmma_bf16(a0, b0, acc[0][0]);
    acc[0][1] = wmma_bf16(a0, b1, acc[0][1]);
    acc[1][0] = wmma_bf16(a1, b0, acc[1][0]);
    acc[1][1] = wmma_bf16(a1, b1, acc[1][1]);
  }

  int rhalf = (lane >> 4) << 3;
#pragma unroll
  for (int i = 0; i < 2; ++i) {
#pragma unroll
    for (int j = 0; j < 2; ++j) {
      int cn = n0 + j * 16 + (lane & 15);
      float bv = bias[cn];
#pragma unroll
      for (int t = 0; t < 8; ++t) {
        int rm = m0 + i * 16 + rhalf + t;
        float v = acc[i][j][t] + bv;
        if (mode == 0) v *= 0.125f;                 // 1/sqrt(HDIM)
        if (mode <= 2) {
          int l = rm >> 2, n = rm & 3;
          int h = cn >> 6, hd = cn & 63;
          long idx;
          if (mode == 2) idx = ((long)(n * NH + h) * HDIM + hd) * LSEQ + l;
          else           idx = ((long)(n * NH + h) * LSEQ + l) * HDIM + hd;
          dQK[idx] = f2bf(v);
        } else {
          dF[(long)rm * EMB + cn] = v;
        }
      }
    }
  }
}

// ---------------------------------------------------------------- attention
// Block = 8 waves, one (batch,head). Waves share K/V tiles staged in LDS via
// async global->LDS copies (double-buffered). Wave w owns 16 query rows.
// Per 32-key chunk: S = Q K^T (4 WMMA), online softmax, O += P V (4 WMMA).
__global__ __launch_bounds__(256) void attn(
    const unsigned short* __restrict__ Qh,   // [BH][LSEQ][HDIM]
    const unsigned short* __restrict__ Kh,   // [BH][LSEQ][HDIM]
    const unsigned short* __restrict__ Vt,   // [BH][HDIM][LSEQ]
    unsigned short* __restrict__ Ao) {       // [LSEQ][NB][EMB] bf16
  // kv[buf]: [0..2047]  K tile, 32 rows(s) x 64 cols(hd)
  //          [2048..]   V tile, 64 rows(hd) x 32 cols(s)
  __shared__ unsigned short kv[2][4096];
  __shared__ unsigned short plds[8][512];    // per-wave 2x16x16 bf16 P tile

  int tid  = threadIdx.x;
  int wid  = tid >> 5;
  int lane = tid & 31;
  int bh = blockIdx.x;
  int m0 = blockIdx.y * 128 + wid * 16;

  const unsigned short* Qb = Qh + (long)bh * LSEQ * HDIM;
  const unsigned short* Kb = Kh + (long)bh * LSEQ * HDIM;
  const unsigned short* Vb = Vt + (long)bh * HDIM * LSEQ;

  unsigned kv0 = (unsigned)(unsigned long long)(&kv[0][0]);  // LDS byte offset
  // K tile: 4096 contiguous bytes; thread copies 16B at byte offset tid*16.
  // V tile: 64 rows, 64B each, global row stride LSEQ*2 bytes.
  int vrow = tid >> 2;            // 0..63  (hd)
  int vseg = (tid & 3) << 3;      // element offset within row: 0,8,16,24

  // issue prefetch of chunk s0 into buffer b
  auto prefetch = [&](int b, int s0) {
    unsigned base = kv0 + (unsigned)b * 8192u;
    async_ld_b128(base + (unsigned)tid * 16u,
                  Kb + (long)s0 * HDIM + tid * 8);
    async_ld_b128(base + 4096u + ((unsigned)vrow * 32u + vseg) * 2u,
                  Vb + (long)vrow * LSEQ + s0 + vseg);
  };

  bf16x16 qa0 = load_frag_g(Qb, m0, HDIM, 0);
  bf16x16 qa1 = load_frag_g(Qb, m0, HDIM, 32);

  f32x8 o[4];
  o[0] = {}; o[1] = {}; o[2] = {}; o[3] = {};
  float mrow[8], lrow[8];
#pragma unroll
  for (int i = 0; i < 8; ++i) { mrow[i] = -3.402823466e38f; lrow[i] = 0.f; }

  unsigned short* myp = plds[wid];
  int koffE = (lane >> 4) << 3;

  prefetch(0, 0);

  for (int s0 = 0; s0 < LSEQ; s0 += 32) {
    int buf = (s0 >> 5) & 1;
    if (s0 + 32 < LSEQ) {
      prefetch(buf ^ 1, s0 + 32);
      asm volatile("s_wait_asynccnt 2" ::: "memory");   // current tiles done
    } else {
      asm volatile("s_wait_asynccnt 0" ::: "memory");
    }
    __syncthreads();                                    // visible to all waves

    const unsigned short* Kt = &kv[buf][0];     // 32 x 64
    const unsigned short* Vtl = &kv[buf][2048]; // 64 x 32

    bf16x16 kb00 = load_frag_g(Kt, 0,  HDIM, 0);
    bf16x16 kb01 = load_frag_g(Kt, 0,  HDIM, 32);
    bf16x16 kb10 = load_frag_g(Kt, 16, HDIM, 0);
    bf16x16 kb11 = load_frag_g(Kt, 16, HDIM, 32);

    f32x8 st0 = {}, st1 = {};
    st0 = wmma_bf16(qa0, kb00, st0);
    st0 = wmma_bf16(qa1, kb01, st0);
    st1 = wmma_bf16(qa0, kb10, st1);
    st1 = wmma_bf16(qa1, kb11, st1);

#pragma unroll
    for (int i = 0; i < 8; ++i) {
      float x = fmaxf(st0[i], st1[i]);
      x = fmaxf(x, __shfl_xor(x, 1, 32));
      x = fmaxf(x, __shfl_xor(x, 2, 32));
      x = fmaxf(x, __shfl_xor(x, 4, 32));
      x = fmaxf(x, __shfl_xor(x, 8, 32));
      float nm = fmaxf(mrow[i], x);
      float sc = __expf(mrow[i] - nm);
      float p0 = __expf(st0[i] - nm);
      float p1 = __expf(st1[i] - nm);
      float rs = p0 + p1;
      rs += __shfl_xor(rs, 1, 32);
      rs += __shfl_xor(rs, 2, 32);
      rs += __shfl_xor(rs, 4, 32);
      rs += __shfl_xor(rs, 8, 32);
      lrow[i] = lrow[i] * sc + rs;
      mrow[i] = nm;
      o[0][i] *= sc; o[1][i] *= sc; o[2][i] *= sc; o[3][i] *= sc;
      int r = i + ((lane >> 4) << 3);
      int c = lane & 15;
      myp[r * 16 + c]       = f2bf(p0);   // key tile 0
      myp[256 + r * 16 + c] = f2bf(p1);   // key tile 1
    }
    asm volatile("" ::: "memory");        // LDS is in-order within a wave
    FragU pf;
    pf.q[0] = *reinterpret_cast<const uint4*>(&myp[(lane & 15) * 16 + koffE]);
    pf.q[1] = *reinterpret_cast<const uint4*>(&myp[256 + (lane & 15) * 16 + koffE]);
    bf16x16 pa = pf.v;
#pragma unroll
    for (int j = 0; j < 4; ++j) {
      bf16x16 vb = load_frag_g(Vtl, j * 16, 32, 0);
      o[j] = wmma_bf16(pa, vb, o[j]);
    }
    __syncthreads();                      // before buf is overwritten
  }

  int n = bh >> 4, h = bh & 15;
#pragma unroll
  for (int j = 0; j < 4; ++j) {
    int e = h * HDIM + j * 16 + (lane & 15);
#pragma unroll
    for (int t = 0; t < 8; ++t) {
      int l = m0 + t + ((lane >> 4) << 3);
      float v = o[j][t] / lrow[t];
      Ao[((long)l * NB + n) * EMB + e] = f2bf(v);
    }
  }
}

// ---------------------------------------------------------------- launch
extern "C" void kernel_launch(void* const* d_in, const int* in_sizes, int n_in,
                              void* d_out, int out_size, void* d_ws, size_t ws_size,
                              hipStream_t stream) {
  const float* q_in   = (const float*)d_in[0];
  const float* k_in   = (const float*)d_in[1];
  const float* v_in   = (const float*)d_in[2];
  const float* w_in   = (const float*)d_in[3];   // (3E, E)
  const float* b_in   = (const float*)d_in[4];   // (3E,)
  const float* w_out  = (const float*)d_in[5];   // (E, E)
  const float* b_out  = (const float*)d_in[6];   // (E,)
  float* out = (float*)d_out;

  const long XQ = (long)MROWS * EMB;       // 8388608
  const long WI = 3L * EMB * EMB;          // 3145728
  const long WO = (long)EMB * EMB;         // 1048576
  const long QH = (long)BH * LSEQ * HDIM;  // 8388608

  unsigned short* ws = (unsigned short*)d_ws;
  unsigned short* Xq = ws;
  unsigned short* Xk = Xq + XQ;
  unsigned short* Xv = Xk + XQ;
  unsigned short* Wi = Xv + XQ;
  unsigned short* Wo = Wi + WI;
  unsigned short* Qh = Wo + WO;
  unsigned short* Kh = Qh + QH;
  unsigned short* Vt = Kh + QH;
  unsigned short* Ao = Vt + QH;

  cvt_f32_bf16<<<4096, 256, 0, stream>>>(q_in,  Xq, (int)XQ);
  cvt_f32_bf16<<<4096, 256, 0, stream>>>(k_in,  Xk, (int)XQ);
  cvt_f32_bf16<<<4096, 256, 0, stream>>>(v_in,  Xv, (int)XQ);
  cvt_f32_bf16<<<4096, 256, 0, stream>>>(w_in,  Wi, (int)WI);
  cvt_f32_bf16<<<4096, 256, 0, stream>>>(w_out, Wo, (int)WO);

  dim3 gg(MROWS / 64, EMB / 128);
  gemm_nt<<<gg, 256, 0, stream>>>(Xq, Wi,                  b_in,           Qh, nullptr, 0);
  gemm_nt<<<gg, 256, 0, stream>>>(Xk, Wi + (long)EMB*EMB,  b_in + EMB,     Kh, nullptr, 1);
  gemm_nt<<<gg, 256, 0, stream>>>(Xv, Wi + 2L*EMB*EMB,     b_in + 2*EMB,   Vt, nullptr, 2);

  attn<<<dim3(BH, LSEQ / 128), 256, 0, stream>>>(Qh, Kh, Vt, Ao);

  gemm_nt<<<gg, 256, 0, stream>>>(Ao, Wo, b_out, nullptr, out, 3);
}
